// WrappedModel_45088566674044
// MI455X (gfx1250) — compile-verified, matrix operand order checked by reading, not run on hardware
//
#include <hip/hip_runtime.h>
#include <stdint.h>

// MI455X / gfx1250, wave32. Pipeline:
//   prep:      embedding f32 -> bf16 (ws), sq_e[256], zero replicated row sums
//   gemm_exp:  cross = A@B^T via v_wmma_f32_16x16x32_bf16 (B cvt'd on the fly,
//              |b|^2 accumulated during cvt), writes exp(-dist/sqrt(512)) to
//              d_out, row partial sums -> 64 replicated accumulators
//   reduce:    64x256 -> inv_sum[256]
//   normalize: d_out *= inv_sum[row]  (float4 vectorized)
//
// ws usage: 256KB (A bf16) + 1KB (sq_e) + 64KB (rep sums) + 1KB (inv) ~ 330KB.

typedef __bf16 bf16_t;
typedef bf16_t v16bf __attribute__((ext_vector_type(16)));
typedef float  v8f   __attribute__((ext_vector_type(8)));

#define EMB   512
#define NEMB  200000
#define BATCH 256
#define NBLK  64      // output columns per block
#define REPL  64      // replicated row-sum buffers (atomic contention relief)

union FragBF {
  v16bf v;
  uint4 q[2];
};

// Convert 8 f32 -> bf16 frag elements [base..base+7], accumulate sum of squares.
__device__ __forceinline__ void cvt_half(FragBF& f, int base, const float4& a,
                                         const float4& b, float& s) {
  f.v[base + 0] = (bf16_t)a.x; f.v[base + 1] = (bf16_t)a.y;
  f.v[base + 2] = (bf16_t)a.z; f.v[base + 3] = (bf16_t)a.w;
  f.v[base + 4] = (bf16_t)b.x; f.v[base + 5] = (bf16_t)b.y;
  f.v[base + 6] = (bf16_t)b.z; f.v[base + 7] = (bf16_t)b.w;
  s += a.x * a.x + a.y * a.y + a.z * a.z + a.w * a.w;
  s += b.x * b.x + b.y * b.y + b.z * b.z + b.w * b.w;
}

// ---------------------------------------------------------------------------
// prep: blocks 0..255 -> bf16-convert one embedding row + sq_e; block 256 ->
// zero the replicated row-sum buffers.
__global__ __launch_bounds__(256) void prep_kernel(
    const float* __restrict__ emb, uint32_t* __restrict__ wsA,
    float* __restrict__ sq_e, float* __restrict__ rep) {
  __shared__ float red[256];
  const int t = threadIdx.x;
  const int b = blockIdx.x;
  if (b < BATCH) {
    float2 e = ((const float2*)(emb + (size_t)b * EMB))[t];
    unsigned short lo = __builtin_bit_cast(unsigned short, (bf16_t)e.x);
    unsigned short hi = __builtin_bit_cast(unsigned short, (bf16_t)e.y);
    wsA[(size_t)b * (EMB / 2) + t] = (uint32_t)lo | ((uint32_t)hi << 16);
    red[t] = e.x * e.x + e.y * e.y;
    __syncthreads();
    for (int off = 128; off > 0; off >>= 1) {
      if (t < off) red[t] += red[t + off];
      __syncthreads();
    }
    if (t == 0) sq_e[b] = red[0];
  } else {
    for (int i = t; i < REPL * BATCH; i += 256) rep[i] = 0.0f;
  }
}

// ---------------------------------------------------------------------------
// gemm_exp: one block = full M=256 x N=64 tile over K=512.
// 8 waves in a 4(M) x 2(N) arrangement; each wave owns 4x2 16x16 acc tiles.
__global__ __launch_bounds__(256) void gemm_exp_kernel(
    const float* __restrict__ orig,      // [NEMB, EMB] f32
    const uint32_t* __restrict__ wsA,    // [BATCH, EMB] bf16 (packed pairs)
    const float* __restrict__ sq_e,      // [BATCH]
    float* __restrict__ rep,             // [REPL, BATCH] row-sum accumulators
    float* __restrict__ out) {           // [BATCH, NEMB] exp(logit) (unnormalized)
  __shared__ float lds_part[BATCH][2];

  const int tid  = threadIdx.x;
  const int lane = tid & 31;
  const int wave = tid >> 5;
  const int mi   = wave >> 1;        // 0..3  (64 M-rows each)
  const int ni   = wave & 1;         // 0..1  (32 N-cols each)
  const int l16  = lane & 15;
  const int lh   = (lane >> 4) & 1;  // K-chunk half selector per WMMA layout

  const int nblock = blockIdx.x * NBLK;

  v8f acc[4][2];
#pragma unroll
  for (int mt = 0; mt < 4; ++mt)
#pragma unroll
    for (int nt = 0; nt < 2; ++nt)
      acc[mt][nt] = (v8f){0.f, 0.f, 0.f, 0.f, 0.f, 0.f, 0.f, 0.f};

  float sqb[2] = {0.0f, 0.0f};       // per-lane partial |b|^2 per N-tile

  const int rowB0 = nblock + ni * 32 + l16;  // B-frag source row (nt adds 16)
  const int rowA0 = mi * 64 + l16;           // A-frag source row (mt adds 16)

#pragma unroll
  for (int kc = 0; kc < EMB / 32; ++kc) {
    const int k0 = kc * 32 + lh * 8;  // lane's first K chunk; second at k0+16

    // B fragments: lane = column n, contiguous K from orig_embs row n.
    FragBF fb[2];
#pragma unroll
    for (int nt = 0; nt < 2; ++nt) {
      const float* pb = orig + (size_t)(rowB0 + nt * 16) * EMB + k0;
      float4 c0 = ((const float4*)pb)[0];
      float4 c1 = ((const float4*)pb)[1];
      float4 c2 = ((const float4*)pb)[4];   // +16 floats
      float4 c3 = ((const float4*)pb)[5];
      cvt_half(fb[nt], 0, c0, c1, sqb[nt]);
      cvt_half(fb[nt], 8, c2, c3, sqb[nt]);
    }

    // All four A fragments first (distinct registers -> partial loadcnt waits
    // instead of full drains before each WMMA), then the 8-WMMA burst.
    FragBF fa[4];
#pragma unroll
    for (int mt = 0; mt < 4; ++mt) {
      const uint32_t* pa =
          wsA + (size_t)(rowA0 + mt * 16) * (EMB / 2) + (k0 >> 1);
      fa[mt].q[0] = ((const uint4*)pa)[0];
      fa[mt].q[1] = ((const uint4*)pa)[2];  // +16 bf16 = +8 dwords
    }
#pragma unroll
    for (int mt = 0; mt < 4; ++mt) {
      acc[mt][0] = __builtin_amdgcn_wmma_f32_16x16x32_bf16(
          false, fa[mt].v, false, fb[0].v, (short)0, acc[mt][0], false, false);
      acc[mt][1] = __builtin_amdgcn_wmma_f32_16x16x32_bf16(
          false, fa[mt].v, false, fb[1].v, (short)0, acc[mt][1], false, false);
    }
  }

  // Combine |b|^2 K-halves (partner lane = lane ^ 16 holds the other chunks).
  float sqbt[2];
#pragma unroll
  for (int nt = 0; nt < 2; ++nt)
    sqbt[nt] = sqb[nt] + __shfl_xor(sqb[nt], 16, 32);

  // Batch the sq_e broadcast loads so their latency overlaps.
  float se[4][8];
#pragma unroll
  for (int mt = 0; mt < 4; ++mt)
#pragma unroll
    for (int r = 0; r < 8; ++r)
      se[mt][r] = sq_e[mi * 64 + mt * 16 + r + lh * 8];

  // exp(-sqrt(d2)/sqrt(512)) == exp2(-sqrt(d2) * log2(e)/sqrt(512)).
  constexpr float KEXP = (float)(1.4426950408889634 / 22.627416997969522);

#pragma unroll
  for (int mt = 0; mt < 4; ++mt) {
#pragma unroll
    for (int r = 0; r < 8; ++r) {
      // C/D layout: VGPR r -> M = r (lanes 0-15) or r+8 (lanes 16-31); N = l16.
      const int m = mi * 64 + mt * 16 + r + lh * 8;
      float esum = 0.0f;
#pragma unroll
      for (int nt = 0; nt < 2; ++nt) {
        float cr = acc[mt][nt][r];
        float d2 = fmaxf(se[mt][r] + sqbt[nt] - 2.0f * cr, 0.0f);
        float E = exp2f(-__fsqrt_rn(d2) * KEXP);
        const int col = nblock + ni * 32 + nt * 16 + l16;
        out[(size_t)m * NEMB + col] = E;
        esum += E;
      }
      // Reduce over the 16 columns of each lane half.
      esum += __shfl_xor(esum, 1, 32);
      esum += __shfl_xor(esum, 2, 32);
      esum += __shfl_xor(esum, 4, 32);
      esum += __shfl_xor(esum, 8, 32);
      if (l16 == 0) lds_part[m][ni] = esum;  // unique (m, ni) per writer
    }
  }
  __syncthreads();

  // One f32 atomic per row per block, spread over REPL replicated buffers.
  {
    float s = lds_part[tid][0] + lds_part[tid][1];
    __hip_atomic_fetch_add(&rep[(blockIdx.x & (REPL - 1)) * BATCH + tid], s,
                           __ATOMIC_RELAXED, __HIP_MEMORY_SCOPE_AGENT);
  }
}

// ---------------------------------------------------------------------------
__global__ __launch_bounds__(256) void reduce_sums_kernel(
    const float* __restrict__ rep, float* __restrict__ inv_sum) {
  const int t = threadIdx.x;
  float s = 0.0f;
  for (int j = 0; j < REPL; ++j) s += rep[j * BATCH + t];
  inv_sum[t] = 1.0f / s;
}

// normalize: float4 over one row chunk per block; NEMB % 4 == 0 and the row
// stride (800000 B) is 16B-aligned, so b128 accesses are legal and coalesced.
#define NVEC (NEMB / 4)  // 50000 float4 per row
__global__ __launch_bounds__(256) void normalize_kernel(
    float4* __restrict__ out4, const float* __restrict__ inv_sum) {
  const int row = blockIdx.y;
  const float inv = inv_sum[row];
  float4* p = out4 + (size_t)row * NVEC;
  const int base = blockIdx.x * 1024 + threadIdx.x;
#pragma unroll
  for (int i = 0; i < 4; ++i) {
    int idx = base + i * 256;
    if (idx < NVEC) {
      float4 v = p[idx];
      v.x *= inv; v.y *= inv; v.z *= inv; v.w *= inv;
      p[idx] = v;
    }
  }
}

// ---------------------------------------------------------------------------
extern "C" void kernel_launch(void* const* d_in, const int* in_sizes, int n_in,
                              void* d_out, int out_size, void* d_ws,
                              size_t ws_size, hipStream_t stream) {
  (void)in_sizes; (void)n_in; (void)out_size; (void)ws_size;
  const float* emb  = (const float*)d_in[0];  // [256, 512] f32
  const float* orig = (const float*)d_in[1];  // [200000, 512] f32
  float* out = (float*)d_out;                 // [256, 200000] f32

  char* ws = (char*)d_ws;
  uint32_t* wsA  = (uint32_t*)ws;                          // 262144 B
  float* sq_e    = (float*)(ws + 262144);                  // 1024 B
  float* rep     = (float*)(ws + 263168);                  // 65536 B
  float* inv_sum = (float*)(ws + 263168 + REPL * BATCH * 4);

  prep_kernel<<<BATCH + 1, 256, 0, stream>>>(emb, wsA, sq_e, rep);
  gemm_exp_kernel<<<NEMB / NBLK, 256, 0, stream>>>(orig, wsA, sq_e, rep, out);
  reduce_sums_kernel<<<1, 256, 0, stream>>>(rep, inv_sum);
  dim3 g((NVEC + 1023) / 1024, BATCH);
  normalize_kernel<<<g, 256, 0, stream>>>((float4*)out, inv_sum);
}
